// AqtDotGeneral_60069412602522
// MI455X (gfx1250) — 8-bit matrix operand order repaired and checked by reading, compiled, not run on hardware
//
#include <hip/hip_runtime.h>
#include <math.h>

typedef __attribute__((ext_vector_type(8))) int v8i;
typedef int v4i32 __attribute__((vector_size(16)));
typedef __attribute__((address_space(1))) v4i32 as1_v4i;
typedef __attribute__((address_space(3))) v4i32 as3_v4i;

#ifndef __has_builtin
#define __has_builtin(x) 0
#endif

#if __has_builtin(__builtin_amdgcn_global_load_async_to_lds_b128)
#define HAVE_ASYNC_LDS 1
#else
#define HAVE_ASYNC_LDS 0
#endif

#if HAVE_ASYNC_LDS
#if __has_builtin(__builtin_amdgcn_s_wait_asynccnt)
#define WAIT_ASYNC(n) __builtin_amdgcn_s_wait_asynccnt(n)
#else
#define WAIT_ASYNC(n) asm volatile("s_wait_asynccnt %0" ::"n"(n))
#endif
#else
#define WAIT_ASYNC(n) ((void)0)
#endif

#define BM 128
#define BN 128
#define BK 64

__global__ void rowmax_scale_kernel(const float* __restrict__ lhs,
                                    float* __restrict__ s_lhs, int K) {
  __shared__ float red[256];
  const int row = blockIdx.x;
  const float* p = lhs + (size_t)row * K;
  float m = 0.f;
  for (int k = threadIdx.x; k < K; k += 256) m = fmaxf(m, fabsf(p[k]));
  red[threadIdx.x] = m;
  __syncthreads();
  for (int s = 128; s > 0; s >>= 1) {
    if (threadIdx.x < s) red[threadIdx.x] = fmaxf(red[threadIdx.x], red[threadIdx.x + s]);
    __syncthreads();
  }
  if (threadIdx.x == 0) {
    const float amax = red[0];
    s_lhs[row] = (amax > 0.f) ? (amax / 127.f) : 1.f;
  }
}

__global__ void colmax_scale_kernel(const float* __restrict__ rhs,
                                    float* __restrict__ s_rhs, int K, int N) {
  const int col = blockIdx.x * blockDim.x + threadIdx.x;
  if (col >= N) return;
  float m = 0.f;
  for (int r = 0; r < K; ++r) m = fmaxf(m, fabsf(rhs[(size_t)r * N + col]));
  s_rhs[col] = (m > 0.f) ? (m / 127.f) : 1.f;
}

__device__ __forceinline__ signed char quant1(float x, float s) {
  return (signed char)(int)rintf(fminf(fmaxf(x / s, -127.f), 127.f));
}

__global__ void quant_lhs_kernel(const float* __restrict__ lhs,
                                 const float* __restrict__ s_lhs,
                                 signed char* __restrict__ qA, int K,
                                 long long total4) {
  const long long i = (long long)blockIdx.x * blockDim.x + threadIdx.x;
  if (i >= total4) return;
  const long long base = i * 4;
  const int row = (int)(base / K);
  const float s = s_lhs[row];
  const float4 v = *(const float4*)(lhs + base);
  char4 o;
  o.x = quant1(v.x, s);
  o.y = quant1(v.y, s);
  o.z = quant1(v.z, s);
  o.w = quant1(v.w, s);
  *(char4*)(qA + base) = o;
}

__global__ void quant_rhsT_kernel(const float* __restrict__ rhs,
                                  const float* __restrict__ s_rhs,
                                  signed char* __restrict__ qBT, int K, int N) {
  const long long i = (long long)blockIdx.x * blockDim.x + threadIdx.x;
  if (i >= (long long)K * N) return;
  const int n = (int)(i % N);
  const int k = (int)(i / N);
  qBT[(size_t)n * K + k] = quant1(rhs[i], s_rhs[n]);
}

__device__ __forceinline__ void stage_tile(const signed char* __restrict__ qA,
                                           const signed char* __restrict__ qBT,
                                           signed char* lA, signed char* lB,
                                           int tid, int m0, int n0, int k0, int K) {
  const signed char* ga[2];
  const signed char* gb[2];
  signed char* la[2];
  signed char* lb[2];
#pragma unroll
  for (int t = 0; t < 2; ++t) {
    const int idx = tid + t * 256;
    const int row = idx >> 2;
    const int seg = (idx & 3) << 4;
    ga[t] = qA + (size_t)(m0 + row) * K + k0 + seg;
    gb[t] = qBT + (size_t)(n0 + row) * K + k0 + seg;
    la[t] = lA + row * BK + seg;
    lb[t] = lB + row * BK + seg;
  }
#if HAVE_ASYNC_LDS
#pragma unroll
  for (int t = 0; t < 2; ++t) {
    __builtin_amdgcn_global_load_async_to_lds_b128((as1_v4i*)ga[t], (as3_v4i*)la[t], 0, 0);
    __builtin_amdgcn_global_load_async_to_lds_b128((as1_v4i*)gb[t], (as3_v4i*)lb[t], 0, 0);
  }
#else
  int4 va[2], vb[2];
#pragma unroll
  for (int t = 0; t < 2; ++t) {
    va[t] = *(const int4*)ga[t];
    vb[t] = *(const int4*)gb[t];
  }
#pragma unroll
  for (int t = 0; t < 2; ++t) {
    *(int4*)la[t] = va[t];
    *(int4*)lb[t] = vb[t];
  }
#endif
}

__global__ __launch_bounds__(256)
void wmma_int8_gemm_kernel(const signed char* __restrict__ qA,
                           const signed char* __restrict__ qBT,
                           const float* __restrict__ sA,
                           const float* __restrict__ sB,
                           float* __restrict__ out, int M, int N, int K) {
  __shared__ __align__(16) signed char As[2][BM * BK];
  __shared__ __align__(16) signed char Bs[2][BN * BK];

  const int tid  = threadIdx.x;
  const int wave = tid >> 5;
  const int lane = tid & 31;
  const int hl   = lane >> 4;
  const int lr   = lane & 15;

  const int m0 = blockIdx.y * BM;
  const int n0 = blockIdx.x * BN;

  const int wm = wave >> 2;
  const int wn = wave & 3;

  v8i acc[4][2] = {};

  const int ntiles = K / BK;

  stage_tile(qA, qBT, As[0], Bs[0], tid, m0, n0, 0, K);

  for (int kt = 0; kt < ntiles; ++kt) {
    const int cur = kt & 1;
    const bool more = (kt + 1 < ntiles);

    if (more) stage_tile(qA, qBT, As[cur ^ 1], Bs[cur ^ 1], tid, m0, n0, (kt + 1) * BK, K);

    if (kt + 2 < ntiles) {
      const int kp = (kt + 2) * BK;
      __builtin_prefetch(&qA[(size_t)(m0 + (tid >> 1)) * K + kp + (tid & 1) * 32], 0, 1);
      __builtin_prefetch(&qBT[(size_t)(n0 + (tid >> 1)) * K + kp + (tid & 1) * 32], 0, 1);
    }

    if (more) WAIT_ASYNC(4);
    else      WAIT_ASYNC(0);
    __syncthreads();

    const signed char* lA = As[cur];
    const signed char* lB = Bs[cur];

    v8i bf[2];
#pragma unroll
    for (int nt = 0; nt < 2; ++nt) {
      const int nrow = (wn * 2 + nt) * 16 + lr;
#pragma unroll
      for (int j = 0; j < 8; ++j) {
        const int kk = (j < 4) ? (hl * 16 + j * 4) : (32 + hl * 16 + (j - 4) * 4);
        bf[nt][j] = *(const int*)(&lB[nrow * BK + kk]);
      }
    }

#pragma unroll
    for (int mt = 0; mt < 4; ++mt) {
      const int arow = (wm * 4 + mt) * 16 + lr;
      v8i af;
#pragma unroll
      for (int j = 0; j < 8; ++j) {
        const int g = j >> 1, p = j & 1;
        const int kk = (j >> 2) * 32 + hl * 16 + (j & 3) * 4;
        af[j] = *(const int*)(&lA[arow * BK + kk]);
      }
#pragma unroll
      for (int nt = 0; nt < 2; ++nt) {
        acc[mt][nt] = __builtin_amdgcn_wmma_i32_16x16x64_iu8(
            true, af, true, bf[nt], acc[mt][nt], false, false);
      }
    }
    __syncthreads();
  }

#pragma unroll
  for (int mt = 0; mt < 4; ++mt) {
#pragma unroll
    for (int nt = 0; nt < 2; ++nt) {
      const int n = n0 + (wn * 2 + nt) * 16 + lr;
      const float sb = sB[n];
#pragma unroll
      for (int v = 0; v < 8; ++v) {
        const int m = m0 + (wm * 4 + mt) * 16 + v + hl * 8;
        out[(size_t)m * N + n] = (float)acc[mt][nt][v] * sA[m] * sb;
      }
    }
  }
}

extern "C" void kernel_launch(void* const* d_in, const int* in_sizes, int n_in,
                              void* d_out, int out_size, void* d_ws, size_t ws_size,
                              hipStream_t stream) {
  (void)n_in;
  (void)ws_size;
  const float* lhs = (const float*)d_in[0];
  const float* rhs = (const float*)d_in[1];
  float* out = (float*)d_out;

  const double szL = (double)in_sizes[0];
  const double szR = (double)in_sizes[1];
  const double szO = (double)out_size;
  const long long K = (long long)(sqrt(szL * szR / szO) + 0.5);
  const long long M = (long long)in_sizes[0] / K;
  const long long N = (long long)in_sizes[1] / K;

  float* s_lhs = (float*)d_ws;
  float* s_rhs = s_lhs + M;
  signed char* qA = (signed char*)(s_rhs + N);
  signed char* qBT = qA + (size_t)M * K;

  rowmax_scale_kernel<<<(int)M, 256, 0, stream>>>(lhs, s_lhs, (int)K);
  colmax_scale_kernel<<<(int)((N + 255) / 256), 256, 0, stream>>>(rhs, s_rhs, (int)K, (int)N);

  const long long total4 = M * K / 4;
  quant_lhs_kernel<<<(int)((total4 + 255) / 256), 256, 0, stream>>>(lhs, s_lhs, qA, (int)K, total4);

  const long long totalR = K * N;
  quant_rhsT_kernel<<<(int)((totalR + 255) / 256), 256, 0, stream>>>(rhs, s_rhs, qBT, (int)K, (int)N);

  dim3 grid((unsigned)(N / BN), (unsigned)(M / BM));
  wmma_int8_gemm_kernel<<<grid, 256, 0, stream>>>(qA, qBT, s_lhs, s_rhs, out,
                                                  (int)M, (int)N, (int)K);
}
